// InitStr_Network_7894149890478
// MI455X (gfx1250) — matrix operand order repaired and checked by reading, not compiled, not run on hardware
//
#include <hip/hip_runtime.h>

// ---------------------------------------------------------------------------
// Problem constants (from reference): B=2, N=128, L=256, dn=64, de=128,
// dh=deh=64, H=4, HD=256, nB=3
// ---------------------------------------------------------------------------
#define BB   2
#define NN   128
#define LL   256
#define DN   64
#define DE   128
#define DH   64
#define HH   4
#define HD   256
#define NEGV (-1e9f)

// ---------------------------------------------------------------------------
// Types / helpers
// ---------------------------------------------------------------------------
typedef __attribute__((ext_vector_type(16))) __bf16 v16bf;
typedef __attribute__((ext_vector_type(8)))  float  v8f;

union ABfrag { v16bf v; unsigned short us[16]; uint4 q[2]; };

__device__ __forceinline__ unsigned short f2bf(float f) {
  unsigned int u = __float_as_uint(f);
  unsigned int r = u + 0x7fffu + ((u >> 16) & 1u);  // round-to-nearest-even
  return (unsigned short)(r >> 16);
}
__device__ __forceinline__ float bf2f(unsigned short h) {
  return __uint_as_float(((unsigned int)h) << 16);
}
__device__ __forceinline__ float eluf(float x) {
  return x > 0.f ? x : (expf(x) - 1.f);
}

__device__ __forceinline__ float waveSum(float v) {
  #pragma unroll
  for (int m = 16; m >= 1; m >>= 1) v += __shfl_xor(v, m, 32);
  return v;
}
__device__ __forceinline__ float waveMax(float v) {
  #pragma unroll
  for (int m = 16; m >= 1; m >>= 1) v = fmaxf(v, __shfl_xor(v, m, 32));
  return v;
}

template <int NW>
__device__ __forceinline__ float blockSum(float v, float* red) {
  int wid = threadIdx.x >> 5, lane = threadIdx.x & 31;
  v = waveSum(v);
  if (lane == 0) red[wid] = v;
  __syncthreads();
  if (wid == 0) {
    float s = (lane < NW) ? red[lane] : 0.f;
    s = waveSum(s);
    if (lane == 0) red[0] = s;
  }
  __syncthreads();
  float r = red[0];
  __syncthreads();
  return r;
}
template <int NW>
__device__ __forceinline__ float blockMax(float v, float* red) {
  int wid = threadIdx.x >> 5, lane = threadIdx.x & 31;
  v = waveMax(v);
  if (lane == 0) red[wid] = v;
  __syncthreads();
  if (wid == 0) {
    float s = (lane < NW) ? red[lane] : -3.4e38f;
    s = waveMax(s);
    if (lane == 0) red[0] = s;
  }
  __syncthreads();
  float r = red[0];
  __syncthreads();
  return r;
}

// ---------------------------------------------------------------------------
// K1: layernorm of msa rows (dn=64) -> f32 msa_n.   One wave per row.
// grid = B*N*L/8 = 8192, block = 256
// ---------------------------------------------------------------------------
__global__ __launch_bounds__(256) void ln_msa_kernel(
    const float* __restrict__ x, const float* __restrict__ g,
    const float* __restrict__ b, float* __restrict__ y) {
  int wid = threadIdx.x >> 5, lane = threadIdx.x & 31;
  size_t row = (size_t)blockIdx.x * 8 + wid;
  const float* xr = x + row * DN;
  float v0 = xr[lane], v1 = xr[lane + 32];
  float mean = waveSum(v0 + v1) * (1.f / 64.f);
  float d0 = v0 - mean, d1 = v1 - mean;
  float var = waveSum(d0 * d0 + d1 * d1) * (1.f / 64.f);
  float rstd = rsqrtf(var + 1e-5f);
  float* yr = y + row * DN;
  yr[lane]      = d0 * rstd * g[lane]      + b[lane];
  yr[lane + 32] = d1 * rstd * g[lane + 32] + b[lane + 32];
}

// ---------------------------------------------------------------------------
// K2: sequence-weight attention + node features + node MLP -> x0 (B,L,64).
// Key factorization: logit[n] = msa_n[n,:]·p + q·bk,  p = wk_s · q  (no k GEMM).
// grid = B*L = 512, block = 128
// ---------------------------------------------------------------------------
__global__ __launch_bounds__(128) void seqweight_kernel(
    const float* __restrict__ msa_n, const float* __restrict__ seq1hot,
    const float* __restrict__ wq_s, const float* __restrict__ bq_s,
    const float* __restrict__ wk_s, const float* __restrict__ bk_s,
    const float* __restrict__ w_ex, const float* __restrict__ b_ex,
    float* __restrict__ x0) {
  __shared__ float mrow[NN * DN];          // 32 KB
  __shared__ float qv[DN], pv[DN], attnS[NN], nfs[DN];
  __shared__ float red[4];
  int bl = blockIdx.x;
  int b = bl >> 8, l = bl & 255;
  int t = threadIdx.x;
  for (int i = t; i < NN * DN; i += 128) {
    int n = i >> 6, d = i & 63;
    mrow[i] = msa_n[(((size_t)b * NN + n) * LL + l) * DN + d];
  }
  __syncthreads();
  if (t < DN) {
    float a = bq_s[t];
    for (int c = 0; c < DN; ++c) a += mrow[c] * wq_s[c * DN + t];
    qv[t] = a * 0.125f;                    // / sqrt(dn)
  }
  __syncthreads();
  float qb = blockSum<4>((t < DN) ? qv[t] * bk_s[t] : 0.f, red);
  if (t < DN) {
    float a = 0.f;
    for (int d = 0; d < DN; ++d) a += wk_s[t * DN + d] * qv[d];
    pv[t] = a;
  }
  __syncthreads();
  float lg = qb;
  for (int c = 0; c < DN; ++c) lg += mrow[t * DN + c] * pv[c];
  float mx = blockMax<4>(lg, red);
  float p  = expf(lg - mx);
  float sm = blockSum<4>(p, red);
  attnS[t] = p / sm;
  __syncthreads();
  if (t < DN) {
    float a = 0.f;
    for (int n = 0; n < NN; ++n) a += attnS[n] * mrow[n * DN + t];
    nfs[t] = a;
  }
  __syncthreads();
  if (t < DH) {
    float a = b_ex[t];
    for (int c = 0; c < DN; ++c) a += nfs[c] * w_ex[c * DH + t];
    for (int s = 0; s < 21; ++s)
      a += seq1hot[(size_t)bl * 21 + s] * w_ex[(DN + s) * DH + t];
    x0[(size_t)bl * DH + t] = eluf(a);
  }
}

// ---------------------------------------------------------------------------
// K3: layernorm of pair rows (de=128) -> bf16 pair_n.  One wave per row.
// grid = B*L*L/8 = 16384, block = 256
// ---------------------------------------------------------------------------
__global__ __launch_bounds__(256) void ln_pair_kernel(
    const float* __restrict__ x, const float* __restrict__ g,
    const float* __restrict__ b, unsigned short* __restrict__ y) {
  int wid = threadIdx.x >> 5, lane = threadIdx.x & 31;
  size_t row = (size_t)blockIdx.x * 8 + wid;
  const float* xr = x + row * DE;
  float v[4], s = 0.f;
  #pragma unroll
  for (int k = 0; k < 4; ++k) { v[k] = xr[lane + 32 * k]; s += v[k]; }
  float mean = waveSum(s) * (1.f / 128.f);
  float vs = 0.f;
  #pragma unroll
  for (int k = 0; k < 4; ++k) { float d = v[k] - mean; vs += d * d; }
  float rstd = rsqrtf(waveSum(vs) * (1.f / 128.f) + 1e-5f);
  unsigned short* yr = y + row * DE;
  #pragma unroll
  for (int k = 0; k < 4; ++k) {
    int c = lane + 32 * k;
    yr[c] = f2bf((v[k] - mean) * rstd * g[c] + b[c]);
  }
}

// ---------------------------------------------------------------------------
// K4: edge MLP  e = elu(pair_n @ w_ee[0:128] + ss * w_ee[128] + b_ee) -> bf16.
// WMMA bf16 16x16x32, K=128 (4 k-steps), per-wave 16x64 output tile.
// w_ee is pre-swizzled in LDS into exact B-fragment layout so each lane's
// 16 operands are one contiguous 32B chunk (2x ds_load_b128, no gathers).
// grid = (B*L*L)/128 = 1024, block = 256 (8 waves, 16 rows each)
// ---------------------------------------------------------------------------
__global__ __launch_bounds__(256) void edge_gemm_kernel(
    const unsigned short* __restrict__ pn,     // (B*L*L, 128) bf16
    const float* __restrict__ w_ee,            // (129, 64) f32
    const float* __restrict__ b_ee,            // (64)
    const int* __restrict__ idx,               // (B, L)
    unsigned short* __restrict__ e_out) {      // (B*L*L, 64) bf16
  // B-fragment layout per ISA: lane l holds N = l%16, K = (l/16)*8 + (j&7) + (j>>3)*16,
  // j = 0..15.  wswz[((s*4+nt)*32 + lane)*16 + j].
  __shared__ __align__(16) unsigned short wswz[4 * 4 * 32 * 16];  // 16 KB
  for (int t = threadIdx.x; t < 4 * 4 * 32 * 16; t += 256) {
    int j    = t & 15;
    int lane = (t >> 4) & 31;
    int nt   = (t >> 9) & 3;
    int s    = t >> 11;
    int k = s * 32 + (lane >> 4) * 8 + (j & 7) + (j >> 3) * 16;
    int n = nt * 16 + (lane & 15);
    wswz[t] = f2bf(w_ee[k * DH + n]);
  }
  __syncthreads();

  int wid = threadIdx.x >> 5, lane = threadIdx.x & 31;
  int r0 = blockIdx.x * 128 + wid * 16;
  int mrow  = r0 + (lane & 15);        // A: M = lane%16
  int khalf = (lane >> 4) * 8;         // A: 8-elem K chunk base per half-wave
  int ncol  = lane & 15;               // C: N = lane%16

  v8f acc[4];
  #pragma unroll
  for (int nt = 0; nt < 4; ++nt)
    #pragma unroll
    for (int r = 0; r < 8; ++r) acc[nt][r] = 0.f;

  #pragma unroll
  for (int s = 0; s < 4; ++s) {                 // K = 4 x 32
    ABfrag a;
    const unsigned short* arow = pn + (size_t)mrow * DE + s * 32;
    a.q[0] = *(const uint4*)(arow + khalf);          // K = kbase..kbase+7
    a.q[1] = *(const uint4*)(arow + khalf + 16);     // K = kbase+16..kbase+23
    #pragma unroll
    for (int nt = 0; nt < 4; ++nt) {
      ABfrag bm;
      const unsigned short* bp = wswz + ((size_t)((s * 4 + nt) * 32 + lane)) * 16;
      bm.q[0] = *(const uint4*)(bp);
      bm.q[1] = *(const uint4*)(bp + 8);
      acc[nt] = __builtin_amdgcn_wmma_f32_16x16x32_bf16(
          false, a.v, false, bm.v, (short)0, acc[nt], false, false);
    }
  }
  // epilogue: + ss*w_ee[128] + b_ee, elu, -> bf16
  #pragma unroll
  for (int nt = 0; nt < 4; ++nt) {
    int n = nt * 16 + ncol;
    float wlast = w_ee[DE * DH + n];
    float bias  = b_ee[n];
    #pragma unroll
    for (int r = 0; r < 8; ++r) {
      int R   = r0 + r + 8 * (lane >> 4);
      int bb  = R >> 16;               // L*L = 65536
      int rem = R & 65535;
      int ii  = rem >> 8, jj = rem & 255;
      float sepf = (float)idx[bb * LL + jj] - (float)idx[bb * LL + ii];
      float sg = (sepf > 0.f) ? 1.f : ((sepf < 0.f) ? -1.f : 0.f);
      float ss = sg * fminf(logf(fabsf(sepf) + 1.f), 5.5f);
      float val = acc[nt][r] + ss * wlast + bias;
      e_out[(size_t)R * DH + n] = f2bf(eluf(val));
    }
  }
}

// ---------------------------------------------------------------------------
// K5 (per block): qk/kk/vv/skip projections of x, plus u = We_h^T qk_h.
// qk/kk stored bf16 in (B,H,L,64) layout for the WMMA logits kernel.
// grid = B*L = 512, block = 256
// ---------------------------------------------------------------------------
__global__ __launch_bounds__(256) void proj_kernel(
    const float* __restrict__ x,
    const float* __restrict__ Wq, const float* __restrict__ bq,
    const float* __restrict__ Wk, const float* __restrict__ bk,
    const float* __restrict__ Wv, const float* __restrict__ bv,
    const float* __restrict__ Wsk, const float* __restrict__ bsk,
    const float* __restrict__ We,
    unsigned short* __restrict__ qk_bf, unsigned short* __restrict__ kk_bf,
    float* __restrict__ vv, float* __restrict__ skip, float* __restrict__ u) {
  __shared__ float xs[DH];
  __shared__ float qks[HD];
  int bl = blockIdx.x;
  int t = threadIdx.x;
  if (t < DH) xs[t] = x[(size_t)bl * DH + t];
  __syncthreads();
  float aq = bq[t], ak = bk[t], av = bv[t], as = bsk[t];
  #pragma unroll 4
  for (int d = 0; d < DH; ++d) {
    float xv = xs[d];
    aq += xv * Wq[d * HD + t];
    ak += xv * Wk[d * HD + t];
    av += xv * Wv[d * HD + t];
    as += xv * Wsk[d * HD + t];
  }
  int b = bl >> 8, l = bl & 255;
  int h = t >> 6, dd = t & 63;
  size_t bhl = ((size_t)(b * HH + h) * LL + l) * 64 + dd;
  qk_bf[bhl] = f2bf(aq);
  kk_bf[bhl] = f2bf(ak);
  vv[(size_t)bl * HD + t]   = av;
  skip[(size_t)bl * HD + t] = as;
  qks[t] = aq;
  __syncthreads();
  // u[b,l,h,c] = sum_d We[c, h*64+d] * qk[h,d]
  float acc = 0.f;
  #pragma unroll 4
  for (int d = 0; d < 64; ++d) acc += We[dd * HD + h * 64 + d] * qks[h * 64 + d];
  u[((size_t)bl * HH + h) * 64 + dd] = acc;
}

// ---------------------------------------------------------------------------
// K6 (per block): logits_qk[b,j,i,h] = sum_d kk[b,i,h,d]*qk[b,j,h,d]  via WMMA.
// Per (b,h): 256x64 @ 64x256 -> 256x256, one 16x16 tile per wave (2 k-steps).
// grid = 256, block = 256  (2048 waves = B*H*16*16 tiles)
// ---------------------------------------------------------------------------
__global__ __launch_bounds__(256) void logits_qk_kernel(
    const unsigned short* __restrict__ kk_bf,   // (B,H,L,64)
    const unsigned short* __restrict__ qk_bf,   // (B,H,L,64)
    float* __restrict__ lqk) {                  // (B, j, i, H)
  int wid = threadIdx.x >> 5, lane = threadIdx.x & 31;
  int w  = blockIdx.x * 8 + wid;
  int bh = w >> 8;
  int tt = w & 255;
  int it = tt >> 4, jt = tt & 15;
  int b = bh >> 2, h = bh & 3;
  const unsigned short* Ab = kk_bf + (size_t)(b * HH + h) * LL * 64;
  const unsigned short* Bb = qk_bf + (size_t)(b * HH + h) * LL * 64;
  int mrow  = it * 16 + (lane & 15);
  int ncol  = jt * 16 + (lane & 15);
  int khalf = (lane >> 4) * 8;
  int kbB   = (lane >> 4) * 8;
  v8f acc;
  #pragma unroll
  for (int r = 0; r < 8; ++r) acc[r] = 0.f;
  #pragma unroll
  for (int s = 0; s < 2; ++s) {                 // K = 64 = 2 x 32
    ABfrag a, bm;
    const unsigned short* ar = Ab + (size_t)mrow * 64 + s * 32;
    a.q[0] = *(const uint4*)(ar + khalf);
    a.q[1] = *(const uint4*)(ar + khalf + 16);
    const unsigned short* br = Bb + (size_t)ncol * 64 + s * 32 + kbB;
    bm.q[0] = *(const uint4*)(br);
    bm.q[1] = *(const uint4*)(br + 16);
    acc = __builtin_amdgcn_wmma_f32_16x16x32_bf16(
        false, a.v, false, bm.v, (short)0, acc, false, false);
  }
  #pragma unroll
  for (int r = 0; r < 8; ++r) {
    int i = it * 16 + r + 8 * (lane >> 4);
    lqk[(((size_t)b * LL + ncol) * LL + i) * HH + h] = acc[r];
  }
}

// ---------------------------------------------------------------------------
// K7 (per block): fused attention per (b, target j).
//   logits = (lqk + e_col·u)/8, mask, softmax over i,
//   msg = alpha^T vv + (alpha^T e_col) @ We_h, + skip, LN, @Wlin, residual ELU.
// Never materializes ee.  grid = B*L = 512, block = 256.
// ---------------------------------------------------------------------------
__global__ __launch_bounds__(256) void attn_kernel(
    const unsigned short* __restrict__ e_bf,   // (B, i, j, 64) bf16
    const float* __restrict__ u,               // (B, L, H, 64)
    const float* __restrict__ lqk,             // (B, j, i, H)
    const float* __restrict__ vv,              // (B, L(i), HD)
    const float* __restrict__ skip,            // (B, L, HD)
    const float* __restrict__ x_in,            // (B, L, 64)
    const int* __restrict__ idx,               // (B, L)
    const float* __restrict__ ln_g, const float* __restrict__ ln_b,
    const float* __restrict__ Wlin, const float* __restrict__ blin,
    const float* __restrict__ We,
    float* __restrict__ x_out) {
  __shared__ __align__(16) unsigned short e_lds[LL * 72];  // padded rows, 36 KB
  __shared__ float u_lds[HH * 64];
  __shared__ float alpha_lds[LL * HH];
  __shared__ float w_lds[HH * 64];
  __shared__ float outn_lds[HD];
  __shared__ float red[8];

  int bj = blockIdx.x;
  int b = bj >> 8, j = bj & 255;
  int t = threadIdx.x;                       // t also indexes source i

  u_lds[t] = u[(size_t)bj * HD + t];
  {
    const unsigned short* er = e_bf + (((size_t)b * LL + t) * LL + j) * DH;
    #pragma unroll
    for (int c = 0; c < DH; c += 8)
      *(uint4*)(e_lds + t * 72 + c) = *(const uint4*)(er + c);
  }
  float4 lq = *(const float4*)(lqk + ((size_t)bj * LL + t) * HH);
  __syncthreads();

  float lg[4];
  {
    float a0 = 0.f, a1 = 0.f, a2 = 0.f, a3 = 0.f;
    const unsigned short* er = e_lds + t * 72;
    #pragma unroll 4
    for (int c = 0; c < DH; ++c) {
      float ev = bf2f(er[c]);
      a0 += ev * u_lds[c];
      a1 += ev * u_lds[64 + c];
      a2 += ev * u_lds[128 + c];
      a3 += ev * u_lds[192 + c];
    }
    float sepf = (float)idx[b * LL + j] - (float)idx[b * LL + t];
    bool masked = !(fabsf(sepf) > 0.f);
    lg[0] = masked ? NEGV : (lq.x + a0) * 0.125f;
    lg[1] = masked ? NEGV : (lq.y + a1) * 0.125f;
    lg[2] = masked ? NEGV : (lq.z + a2) * 0.125f;
    lg[3] = masked ? NEGV : (lq.w + a3) * 0.125f;
  }
  // softmax over i (= t) for each head
  for (int h = 0; h < HH; ++h) {
    float mx = blockMax<8>(lg[h], red);
    float p  = expf(lg[h] - mx);
    float s  = blockSum<8>(p, red);
    alpha_lds[t * HH + h] = p / s;
  }
  __syncthreads();

  int h = t >> 6, c = t & 63;                // (h, c) == (h, d) since hd == t
  float wvv = 0.f, mv = 0.f;
  #pragma unroll 2
  for (int i = 0; i < LL; ++i) {
    float a = alpha_lds[i * HH + h];
    wvv += a * bf2f(e_lds[i * 72 + c]);
    mv  += a * vv[((size_t)b * LL + i) * HD + t];
  }
  w_lds[t] = wvv;
  __syncthreads();

  float msg = mv;
  #pragma unroll 4
  for (int cc = 0; cc < 64; ++cc) msg += w_lds[h * 64 + cc] * We[cc * HD + t];
  float outv = msg + skip[(size_t)bj * HD + t];
  float mean = blockSum<8>(outv, red) * (1.f / 256.f);
  float dv = outv - mean;
  float var = blockSum<8>(dv * dv, red) * (1.f / 256.f);
  outn_lds[t] = dv * rsqrtf(var + 1e-5f) * ln_g[t] + ln_b[t];
  __syncthreads();

  if (t < DH) {
    float a = blin[t];
    for (int hd = 0; hd < HD; ++hd) a += outn_lds[hd] * Wlin[hd * DH + t];
    float xv = a + x_in[(size_t)bj * DH + t];
    x_out[(size_t)bj * DH + t] = eluf(xv);
  }
}

// ---------------------------------------------------------------------------
// K8: xyz head  (B,L,64) @ (64,9) + b -> (B,L,3,3)
// grid = B*L = 512, block = 32
// ---------------------------------------------------------------------------
__global__ __launch_bounds__(32) void xyz_kernel(
    const float* __restrict__ x, const float* __restrict__ w,
    const float* __restrict__ b, float* __restrict__ out) {
  int bl = blockIdx.x;
  int lane = threadIdx.x;
  if (lane < 9) {
    float acc = b[lane];
    for (int d = 0; d < DH; ++d) acc += x[(size_t)bl * DH + d] * w[d * 9 + lane];
    out[(size_t)bl * 9 + lane] = acc;
  }
}

// ---------------------------------------------------------------------------
// Host-side launch
// ---------------------------------------------------------------------------
extern "C" void kernel_launch(void* const* d_in, const int* in_sizes, int n_in,
                              void* d_out, int out_size, void* d_ws, size_t ws_size,
                              hipStream_t stream) {
  (void)in_sizes; (void)n_in; (void)out_size; (void)ws_size;
  const float* seq1hot   = (const float*)d_in[0];
  const float* msa       = (const float*)d_in[1];
  const float* pair      = (const float*)d_in[2];
  const float* ln_node_g = (const float*)d_in[3];
  const float* ln_node_b = (const float*)d_in[4];
  const float* ln_edge_g = (const float*)d_in[5];
  const float* ln_edge_b = (const float*)d_in[6];
  const float* wq_s      = (const float*)d_in[7];
  const float* bq_s      = (const float*)d_in[8];
  const float* wk_s      = (const float*)d_in[9];
  const float* bk_s      = (const float*)d_in[10];
  const float* w_ex      = (const float*)d_in[11];
  const float* b_ex      = (const float*)d_in[12];
  const float* w_ee      = (const float*)d_in[13];
  const float* b_ee      = (const float*)d_in[14];
  const float* Wq        = (const float*)d_in[15];
  const float* bq        = (const float*)d_in[16];
  const float* Wk        = (const float*)d_in[17];
  const float* bk        = (const float*)d_in[18];
  const float* Wv        = (const float*)d_in[19];
  const float* bv        = (const float*)d_in[20];
  const float* We        = (const float*)d_in[21];
  const float* Wskip     = (const float*)d_in[22];
  const float* bskip     = (const float*)d_in[23];
  const float* ln_g      = (const float*)d_in[24];
  const float* ln_b      = (const float*)d_in[25];
  const float* Wlin      = (const float*)d_in[26];
  const float* blin      = (const float*)d_in[27];
  const float* w_xyz     = (const float*)d_in[28];
  const float* b_xyz     = (const float*)d_in[29];
  const int*   idx       = (const int*)d_in[30];
  float* out = (float*)d_out;

  char* ws = (char*)d_ws;
  size_t off = 0;
  auto alloc = [&](size_t bytes) -> void* {
    void* p = ws + off;
    off += (bytes + 255) & ~(size_t)255;
    return p;
  };
  float*          msa_n   = (float*)alloc((size_t)BB * NN * LL * DN * 4);      // 16.8 MB
  unsigned short* pair_n  = (unsigned short*)alloc((size_t)BB * LL * LL * DE * 2); // 33.5 MB
  unsigned short* e_bf    = (unsigned short*)alloc((size_t)BB * LL * LL * DH * 2); // 16.8 MB
  float*          x0      = (float*)alloc((size_t)BB * LL * DH * 4);
  float*          x1      = (float*)alloc((size_t)BB * LL * DH * 4);
  unsigned short* qk_bf   = (unsigned short*)alloc((size_t)BB * HH * LL * 64 * 2);
  unsigned short* kk_bf   = (unsigned short*)alloc((size_t)BB * HH * LL * 64 * 2);
  float*          vvf     = (float*)alloc((size_t)BB * LL * HD * 4);
  float*          skipf   = (float*)alloc((size_t)BB * LL * HD * 4);
  float*          uf      = (float*)alloc((size_t)BB * LL * HH * 64 * 4);
  float*          lqk     = (float*)alloc((size_t)BB * LL * LL * HH * 4);      // 2.1 MB

  ln_msa_kernel<<<(BB * NN * LL) / 8, 256, 0, stream>>>(msa, ln_node_g, ln_node_b, msa_n);
  seqweight_kernel<<<BB * LL, 128, 0, stream>>>(msa_n, seq1hot, wq_s, bq_s, wk_s, bk_s,
                                                w_ex, b_ex, x0);
  ln_pair_kernel<<<(BB * LL * LL) / 8, 256, 0, stream>>>(pair, ln_edge_g, ln_edge_b, pair_n);
  edge_gemm_kernel<<<(BB * LL * LL) / 128, 256, 0, stream>>>(pair_n, w_ee, b_ee, idx, e_bf);

  for (int blk = 0; blk < 3; ++blk) {
    float* x_in  = (blk & 1) ? x1 : x0;
    float* x_out = (blk & 1) ? x0 : x1;
    size_t wofs = (size_t)blk * DH * HD;    // 64*256 per block
    proj_kernel<<<BB * LL, 256, 0, stream>>>(
        x_in,
        Wq + wofs, bq + blk * HD,
        Wk + wofs, bk + blk * HD,
        Wv + wofs, bv + blk * HD,
        Wskip + wofs, bskip + blk * HD,
        We + wofs,
        qk_bf, kk_bf, vvf, skipf, uf);
    logits_qk_kernel<<<256, 256, 0, stream>>>(kk_bf, qk_bf, lqk);
    attn_kernel<<<BB * LL, 256, 0, stream>>>(
        e_bf, uf, lqk, vvf, skipf, x_in, idx,
        ln_g + blk * HD, ln_b + blk * HD,
        Wlin + (size_t)blk * HD * DH, blin + blk * DH,
        We + wofs, x_out);
  }
  // after blk 0->x1, 1->x0, 2->x1 : final x is x1
  xyz_kernel<<<BB * LL, 32, 0, stream>>>(x1, w_xyz, b_xyz, out);
}
